// LISA_33414845563011
// MI455X (gfx1250) — matrix operand order repaired and checked by reading, not compiled, not run on hardware
//
#include <hip/hip_runtime.h>
#include <hip/hip_bf16.h>

typedef __attribute__((ext_vector_type(16))) _Float16 v16h;
typedef __attribute__((ext_vector_type(8)))  _Float16 v8h;
typedef __attribute__((ext_vector_type(4)))  _Float16 v4h;
typedef __attribute__((ext_vector_type(8)))  float    v8f;

#define LISA_B 4
#define LISA_Q 32768
#define LISA_N 8192
#define LISA_D 64
#define ACT_STRIDE 264   // 256 + 8 halves pad (16B-aligned rows, bank-spread)

// ws layout (halves): wt0[256*224] | wt1..wt3[256*256] | wt4[16*256] ; then f32 biases
#define OFF0 0
#define OFF1 (256 * 224)
#define OFF2 (OFF1 + 256 * 256)
#define OFF3 (OFF2 + 256 * 256)
#define OFF4 (OFF3 + 256 * 256)
#define WT_TOTAL (OFF4 + 16 * 256)

__device__ __forceinline__ v16h cat16(v8h lo, v8h hi) {
  v16h r;
#pragma unroll
  for (int i = 0; i < 8; ++i) { r[i] = lo[i]; r[8 + i] = hi[i]; }
  return r;
}

// ---------------- weight prep: transpose + pad + f16 convert -----------------
__global__ void prep_w(const float* __restrict__ W, const float* __restrict__ bsrc,
                       _Float16* __restrict__ wt, float* __restrict__ bt,
                       int Kpad, int Kin, int Nsrc, int Npad, int layer0) {
  const int t = blockIdx.x * blockDim.x + threadIdx.x;
  const int total = Npad * Kpad;
  if (t < total) {
    const int n = t / Kpad;
    const int k = t - n * Kpad;
    float v = 0.0f;
    if (n < Nsrc) {
      int ks = -1;
      if (layer0) {               // input order: feat[0..191], PE[192..204], pad
        if (k < 192) ks = k + 13; // reference order: [PE(13), feat(192)]
        else if (k < 205) ks = k - 192;
      } else if (k < Kin) {
        ks = k;
      }
      if (ks >= 0) v = W[(size_t)ks * Nsrc + n];
    }
    wt[t] = (_Float16)v;
  }
  if (t < Npad) bt[t] = (t < Nsrc) ? bsrc[t] : 0.0f;
}

// -------- one MLP layer on a wave-private 32-row tile (2 WMMAs per B load) ---
template <int KT, bool RELU>
__device__ __forceinline__ void mlp_layer(_Float16* act, const _Float16* __restrict__ wt,
                                          const float* __restrict__ bias,
                                          const _Float16* nextw, int lane) {
  const int row   = lane & 15;
  const int half  = lane >> 4;      // 0 or 1
  const int koffA = half * 8;       // A-matrix 16-bit layout: hi half-wave owns K+8 / K+24
  const int koffB = half * 8;       // B-matrix 16-bit layout: hi half-wave owns K+8 / K+24
  const int rbase = half * 8;       // C/D layout: hi half-wave holds rows M=8..15

  v16h A0[KT], A1[KT];              // two 16-row tiles resident in VGPRs
#pragma unroll
  for (int kt = 0; kt < KT; ++kt) {
    const _Float16* p0 = act + row * ACT_STRIDE + kt * 32 + koffA;
    const _Float16* p1 = p0 + 16 * ACT_STRIDE;
    A0[kt] = cat16(*(const v8h*)p0, *(const v8h*)(p0 + 16));
    A1[kt] = cat16(*(const v8h*)p1, *(const v8h*)(p1 + 16));
  }
  asm volatile("" ::: "memory");    // all A reads precede the in-place output stores

  for (int nt = 0; nt < 16; ++nt) {
    const int col = nt * 16 + row;
    const _Float16* wp = wt + col * (KT * 32) + koffB;
    v8f acc0 = {0.f, 0.f, 0.f, 0.f, 0.f, 0.f, 0.f, 0.f};
    v8f acc1 = {0.f, 0.f, 0.f, 0.f, 0.f, 0.f, 0.f, 0.f};
#pragma unroll
    for (int kt = 0; kt < KT; ++kt) {
      v16h Bt = cat16(*(const v8h*)(wp + kt * 32), *(const v8h*)(wp + kt * 32 + 16));
      acc0 = __builtin_amdgcn_wmma_f32_16x16x32_f16(false, A0[kt], false, Bt,
                                                    (short)0, acc0, false, false);
      acc1 = __builtin_amdgcn_wmma_f32_16x16x32_f16(false, A1[kt], false, Bt,
                                                    (short)0, acc1, false, false);
    }
    if (nextw) __builtin_prefetch(nextw + ((size_t)(nt * 32 + lane) << 6), 0, 3);
    const float bv = bias[col];
#pragma unroll
    for (int j = 0; j < 8; ++j) {
      float v0 = acc0[j] + bv;
      float v1 = acc1[j] + bv;
      if (RELU) { v0 = fmaxf(v0, 0.0f); v1 = fmaxf(v1, 0.0f); }
      act[(rbase + j) * ACT_STRIDE + col]        = (_Float16)v0;
      act[(16 + rbase + j) * ACT_STRIDE + col]   = (_Float16)v1;
    }
  }
  asm volatile("" ::: "memory");
}

__device__ __forceinline__ void mlp_final(_Float16* act, const _Float16* __restrict__ wt,
                                          const float* __restrict__ bias,
                                          float* pred, int lane) {
  const int row   = lane & 15;
  const int half  = lane >> 4;
  const int koffA = half * 8;
  const int koffB = half * 8;
  const int rbase = half * 8;

  v16h A0[8], A1[8];
#pragma unroll
  for (int kt = 0; kt < 8; ++kt) {
    const _Float16* p0 = act + row * ACT_STRIDE + kt * 32 + koffA;
    const _Float16* p1 = p0 + 16 * ACT_STRIDE;
    A0[kt] = cat16(*(const v8h*)p0, *(const v8h*)(p0 + 16));
    A1[kt] = cat16(*(const v8h*)p1, *(const v8h*)(p1 + 16));
  }
  asm volatile("" ::: "memory");

  const _Float16* wp = wt + row * 256 + koffB;  // n-tile 0 only (out padded 1->16)
  v8f acc0 = {0.f, 0.f, 0.f, 0.f, 0.f, 0.f, 0.f, 0.f};
  v8f acc1 = {0.f, 0.f, 0.f, 0.f, 0.f, 0.f, 0.f, 0.f};
#pragma unroll
  for (int kt = 0; kt < 8; ++kt) {
    v16h Bt = cat16(*(const v8h*)(wp + kt * 32), *(const v8h*)(wp + kt * 32 + 16));
    acc0 = __builtin_amdgcn_wmma_f32_16x16x32_f16(false, A0[kt], false, Bt,
                                                  (short)0, acc0, false, false);
    acc1 = __builtin_amdgcn_wmma_f32_16x16x32_f16(false, A1[kt], false, Bt,
                                                  (short)0, acc1, false, false);
  }
  if (row == 0) {                   // column 0 of D holds the prediction
#pragma unroll
    for (int j = 0; j < 8; ++j) {
      pred[rbase + j]      = acc0[j] + bias[0];
      pred[16 + rbase + j] = acc1[j] + bias[0];
    }
  }
  asm volatile("" ::: "memory");
}

// ------------------------------- fused kernel --------------------------------
__global__ __launch_bounds__(64) void lisa_fused(const float* __restrict__ coord,
                                                 const float* __restrict__ latent,
                                                 const _Float16* __restrict__ wbase,
                                                 const float* __restrict__ bbase,
                                                 float* __restrict__ out) {
  __shared__ __align__(16) _Float16 sAct[2][32][ACT_STRIDE];
  __shared__ float sArea[2][32];
  __shared__ float sPred[2][32];

  const int w     = threadIdx.x >> 5;
  const int lane  = threadIdx.x & 31;
  const int tileQ = (blockIdx.x * 2 + w) * 16;   // 16 queries (32 rows) per wave

  // -------- build inputs: 1 lane per (query, ensemble-path) row --------
  {
    const int m    = lane;                       // row in tile, 0..31
    const int qg   = tileQ + (m >> 1);           // global query index, 0..131071
    const int path = m & 1;                      // 0: vx=-2, 1: vx=0
    const int b    = qg >> 15;                   // Q = 32768
    const int qq   = qg & (LISA_Q - 1);
    const float c  = coord[(size_t)b * LISA_Q + qq];

    const float shift = (path == 0 ? (-2.0f / (float)LISA_N) : 0.0f) + 1e-6f;
    float cc = fminf(fmaxf(c + shift, -1.0f + 1e-6f), 1.0f - 1e-6f);
    // uniform-grid searchsorted(left): idx = ceil(((cc+1)*n - 1) / 2)
    int idx = (int)ceilf((cc + 1.0f) * (0.5f * (float)LISA_N) - 0.5f);
    idx = idx < 0 ? 0 : (idx > LISA_N - 1 ? LISA_N - 1 : idx);
    const float qcoord = -1.0f + (2.0f * (float)idx + 1.0f) * (1.0f / (float)LISA_N);
    const float rel = (c - qcoord) * (float)LISA_N;

    _Float16* arow = &sAct[w][m][0];
#pragma unroll
    for (int f = 0; f < 192; f += 4) {           // prev/cur/next latent, 48 x float4
      int ix = idx + (f >> 6) - 1;
      ix = ix < 0 ? 0 : (ix > LISA_N - 1 ? LISA_N - 1 : ix);
      const float4 v = *(const float4*)(latent + (((size_t)b * LISA_N + ix) << 6) + (f & 63));
      v4h hv = {(_Float16)v.x, (_Float16)v.y, (_Float16)v.z, (_Float16)v.w};
      *(v4h*)(arow + f) = hv;
    }
    sArea[w][m] = fabsf(rel) + 1e-9f;
    arow[192] = (_Float16)rel;
#pragma unroll
    for (int k = 0; k < 6; ++k) {
      const float fk = (float)(1 << k) * rel;
      arow[193 + 2 * k] = (_Float16)__sinf(fk);
      arow[194 + 2 * k] = (_Float16)__cosf(fk);
    }
#pragma unroll
    for (int i = 205; i < 224; ++i) arow[i] = (_Float16)0.0f;
  }
  asm volatile("" ::: "memory");   // DS ops are in-order per wave; block compiler reorder

  _Float16* act = &sAct[w][0][0];
  mlp_layer<7, true>(act, wbase + OFF0, bbase + 0,   wbase + OFF1, lane);  // 224 -> 256
  mlp_layer<8, true>(act, wbase + OFF1, bbase + 256, wbase + OFF2, lane);  // 256 -> 256
  mlp_layer<8, true>(act, wbase + OFF2, bbase + 512, wbase + OFF3, lane);
  mlp_layer<8, true>(act, wbase + OFF3, bbase + 768, wbase + OFF4, lane);
  mlp_final(act, wbase + OFF4, bbase + 1024, &sPred[w][0], lane);

  // -------- local ensemble: swap area weights across the two paths --------
  if (lane < 16) {
    const int q = lane;
    const float p0 = sPred[w][2 * q];
    const float p1 = sPred[w][2 * q + 1];
    const float a0 = sArea[w][2 * q];        // |rel| for vx=-2
    const float a1 = sArea[w][2 * q + 1];    // |rel| for vx=0
    const float tot = a0 + a1;
    out[tileQ + q] = p0 * (a1 / tot) + p1 * (a0 / tot);
  }
}

// --------------------------------- launcher ----------------------------------
extern "C" void kernel_launch(void* const* d_in, const int* in_sizes, int n_in,
                              void* d_out, int out_size, void* d_ws, size_t ws_size,
                              hipStream_t stream) {
  (void)in_sizes; (void)n_in; (void)out_size; (void)ws_size;
  const float* coord  = (const float*)d_in[0];
  const float* latent = (const float*)d_in[1];
  const float* W0 = (const float*)d_in[2];  const float* b0 = (const float*)d_in[3];
  const float* W1 = (const float*)d_in[4];  const float* b1 = (const float*)d_in[5];
  const float* W2 = (const float*)d_in[6];  const float* b2 = (const float*)d_in[7];
  const float* W3 = (const float*)d_in[8];  const float* b3 = (const float*)d_in[9];
  const float* W4 = (const float*)d_in[10]; const float* b4 = (const float*)d_in[11];

  _Float16* wt = (_Float16*)d_ws;
  float*    bt = (float*)((char*)d_ws + (size_t)WT_TOTAL * 2);

  prep_w<<<(256 * 224 + 255) / 256, 256, 0, stream>>>(W0, b0, wt + OFF0, bt + 0,    224, 205, 256, 256, 1);
  prep_w<<<(256 * 256 + 255) / 256, 256, 0, stream>>>(W1, b1, wt + OFF1, bt + 256,  256, 256, 256, 256, 0);
  prep_w<<<(256 * 256 + 255) / 256, 256, 0, stream>>>(W2, b2, wt + OFF2, bt + 512,  256, 256, 256, 256, 0);
  prep_w<<<(256 * 256 + 255) / 256, 256, 0, stream>>>(W3, b3, wt + OFF3, bt + 768,  256, 256, 256, 256, 0);
  prep_w<<<(16  * 256 + 255) / 256, 256, 0, stream>>>(W4, b4, wt + OFF4, bt + 1024, 256, 256, 1,   16,  0);

  // 4096 blocks x 2 waves; each wave: 16 queries x 2 ensemble paths = 32 rows
  lisa_fused<<<4096, 64, 0, stream>>>(coord, latent, wt, bt, (float*)d_out);
}
